// InnerProductDecoder_34978213659346
// MI455X (gfx1250) — matrix operand order repaired and checked by reading, not compiled, not run on hardware
//
#include <hip/hip_runtime.h>
#include <hip/hip_bf16.h>

typedef _Float16 v8h  __attribute__((ext_vector_type(8)));
typedef _Float16 v16h __attribute__((ext_vector_type(16)));
typedef float    v8f  __attribute__((ext_vector_type(8)));

__device__ __forceinline__ float fast_sigmoid(float x)
{
    // 1 / (1 + exp(-x)) = 1 / (1 + exp2(-x * log2(e)))
    return __builtin_amdgcn_rcpf(1.0f + __builtin_amdgcn_exp2f(x * -1.44269504088896f));
}

// ---------------------------------------------------------------------------
// Kernel 1: fc1  —  Zh[M,H](f16) = f16( Z[M,K](f32) @ W[H,K]^T + b[H] )
// One wave: 16-row stripe x all H columns. Compile-time tile counts keep the
// 8 x v8f accumulators in VGPRs. Z is streamed once -> non-temporal loads.
// ---------------------------------------------------------------------------
template <int KTILES, int NTILES>
__global__ __launch_bounds__(256)
void fc1_wmma_kernel(const float* __restrict__ Z, const float* __restrict__ W,
                     const float* __restrict__ bias, _Float16* __restrict__ Zh,
                     int M)
{
    constexpr int K = KTILES * 32;
    constexpr int H = NTILES * 16;

    const int lane = threadIdx.x & 31;
    const int wid  = blockIdx.x * (blockDim.x >> 5) + (threadIdx.x >> 5);
    const int rowbase = wid * 16;
    if (rowbase >= M) return;

    const int  col = lane & 15;
    const bool hi  = (lane & 16) != 0;

    const float* zrow = Z + (size_t)(rowbase + col) * K;

    v8f acc[NTILES];
#pragma unroll
    for (int i = 0; i < NTILES; ++i) acc[i] = v8f{};

#pragma unroll
    for (int kt = 0; kt < KTILES; ++kt) {
        const int koff = kt * 32;

        // A fragment: rows of Z (streamed once -> NT loads), f32 -> f16
        const int kbA = koff + (hi ? 16 : 0);
        v16h afrag;
        {
            const float* p = zrow + kbA;
#pragma unroll
            for (int i = 0; i < 8; ++i) {
                afrag[i]     = (_Float16)__builtin_nontemporal_load(p + i);
                afrag[i + 8] = (_Float16)__builtin_nontemporal_load(p + i + 8);
            }
        }

        // B fragments: rows of W (== columns of W^T), tiny + reused -> RT loads
        const int kbB = koff + (hi ? 16 : 0);
#pragma unroll
        for (int nt = 0; nt < NTILES; ++nt) {
            const float* wrow = W + (size_t)(nt * 16 + col) * K + kbB;
            v16h bfrag;
#pragma unroll
            for (int i = 0; i < 16; ++i) bfrag[i] = (_Float16)wrow[i];

            acc[nt] = __builtin_amdgcn_wmma_f32_16x16x32_f16(
                false, afrag, false, bfrag, (short)0, acc[nt], false, false);
        }
    }

    // Epilogue: +bias, f32 -> f16, row-major store (RT: kernel 2 reuses Zh)
    const int rbase = rowbase + (hi ? 8 : 0);
#pragma unroll
    for (int nt = 0; nt < NTILES; ++nt) {
        const int   outcol = nt * 16 + col;
        const float bv     = bias[outcol];
#pragma unroll
        for (int d = 0; d < 8; ++d) {
            Zh[(size_t)(rbase + d) * H + outcol] = (_Float16)(acc[nt][d] + bv);
        }
    }
}

// ---------------------------------------------------------------------------
// Kernel 2: adj[g] = sigmoid( Zh_g @ Zh_g^T ), Zh_g:[N,H] f16 -> out f32.
// One wave computes a 16x64 output tile (A frag reused across 4 N-tiles).
// Zh fragments are L2-resident (16 MB total, reused ~64x) -> RT loads.
// Output is 268 MB write-once -> non-temporal stores so it never evicts Zh
// from the 192 MB L2.
// ---------------------------------------------------------------------------
template <int KTILES>
__global__ __launch_bounds__(256)
void adj_wmma_kernel(const _Float16* __restrict__ Zh, float* __restrict__ out,
                     int G, int N)
{
    constexpr int H = KTILES * 32;

    const int lane = threadIdx.x & 31;
    const int wid  = blockIdx.x * (blockDim.x >> 5) + (threadIdx.x >> 5);

    const int mt_per = N >> 4;           // row tiles per graph
    const int ng_per = N >> 6;           // 64-wide column groups per graph
    const int wpg    = mt_per * ng_per;

    const int g = wid / wpg;
    if (g >= G) return;
    const int r       = wid % wpg;
    const int rowbase = (r / ng_per) * 16;
    const int colbase = (r % ng_per) * 64;

    const int  ln = lane & 15;
    const bool hi = (lane & 16) != 0;

    const _Float16* Zg   = Zh + (size_t)g * N * H;
    const _Float16* arow = Zg + (size_t)(rowbase + ln) * H;
    const _Float16* brow = Zg + (size_t)(colbase + ln) * H;

    v8f acc0 = {}, acc1 = {}, acc2 = {}, acc3 = {};

#pragma unroll
    for (int kt = 0; kt < KTILES; ++kt) {
        const int koff = kt * 32;

        // A fragment: two contiguous 16B chunks per lane
        const int kbA = koff + (hi ? 16 : 0);
        v16h afrag;
        {
            v8h lo = *(const v8h*)(arow + kbA);
            v8h hh = *(const v8h*)(arow + kbA + 8);
#pragma unroll
            for (int i = 0; i < 8; ++i) { afrag[i] = lo[i]; afrag[i + 8] = hh[i]; }
        }

        // B fragments: one contiguous 32B chunk per lane, 4 column tiles
        const int kbB = koff + (hi ? 16 : 0);
        const _Float16* bp = brow + kbB;
        v16h b0 = *(const v16h*)(bp);
        v16h b1 = *(const v16h*)(bp + (size_t)16 * H);
        v16h b2 = *(const v16h*)(bp + (size_t)32 * H);
        v16h b3 = *(const v16h*)(bp + (size_t)48 * H);

        acc0 = __builtin_amdgcn_wmma_f32_16x16x32_f16(false, afrag, false, b0, (short)0, acc0, false, false);
        acc1 = __builtin_amdgcn_wmma_f32_16x16x32_f16(false, afrag, false, b1, (short)0, acc1, false, false);
        acc2 = __builtin_amdgcn_wmma_f32_16x16x32_f16(false, afrag, false, b2, (short)0, acc2, false, false);
        acc3 = __builtin_amdgcn_wmma_f32_16x16x32_f16(false, afrag, false, b3, (short)0, acc3, false, false);
    }

    // Epilogue: sigmoid + coalesced non-temporal fp32 stores
    const size_t obase = (size_t)g * N * N;
    const int    rbase = rowbase + (hi ? 8 : 0);
#pragma unroll
    for (int d = 0; d < 8; ++d) {
        float* rp = out + obase + (size_t)(rbase + d) * N + colbase + ln;
        __builtin_nontemporal_store(fast_sigmoid(acc0[d]), rp);
        __builtin_nontemporal_store(fast_sigmoid(acc1[d]), rp + 16);
        __builtin_nontemporal_store(fast_sigmoid(acc2[d]), rp + 32);
        __builtin_nontemporal_store(fast_sigmoid(acc3[d]), rp + 48);
    }
}

// ---------------------------------------------------------------------------
// Generic fallbacks (runtime shapes) — only used if dims deviate from the
// reference configuration. Scalar, correct, slow.
// ---------------------------------------------------------------------------
__global__ void fc1_generic_kernel(const float* __restrict__ Z, const float* __restrict__ W,
                                   const float* __restrict__ bias, _Float16* __restrict__ Zh,
                                   int M, int K, int H)
{
    int idx = blockIdx.x * blockDim.x + threadIdx.x;
    if (idx >= M * H) return;
    int m = idx / H, h = idx % H;
    float s = bias[h];
    for (int k = 0; k < K; ++k) s += Z[(size_t)m * K + k] * W[(size_t)h * K + k];
    Zh[idx] = (_Float16)s;
}

__global__ void adj_generic_kernel(const _Float16* __restrict__ Zh, float* __restrict__ out,
                                   int G, int N, int H)
{
    size_t idx = (size_t)blockIdx.x * blockDim.x + threadIdx.x;
    size_t total = (size_t)G * N * N;
    if (idx >= total) return;
    size_t g = idx / ((size_t)N * N);
    size_t rem = idx % ((size_t)N * N);
    int i = (int)(rem / N), j = (int)(rem % N);
    const _Float16* a = Zh + (g * N + i) * H;
    const _Float16* b = Zh + (g * N + j) * H;
    float s = 0.f;
    for (int h = 0; h < H; ++h) s += (float)a[h] * (float)b[h];
    out[idx] = fast_sigmoid(s);
}

// ---------------------------------------------------------------------------
extern "C" void kernel_launch(void* const* d_in, const int* in_sizes, int n_in,
                              void* d_out, int out_size, void* d_ws, size_t ws_size,
                              hipStream_t stream)
{
    (void)n_in; (void)out_size; (void)ws_size;

    const float* Z    = (const float*)d_in[0];   // [G*N, K] f32
    const float* W    = (const float*)d_in[1];   // [H, K]   f32
    const float* bias = (const float*)d_in[2];   // [H]      f32
    // d_in[3] = node_slice (uniform graphs; dims derived from sizes)

    const int H = in_sizes[2];          // 128
    const int K = in_sizes[1] / H;      // 256
    const int G = in_sizes[3] - 1;      // 64
    const int M = in_sizes[0] / K;      // 65536 total nodes
    const int N = M / G;                // 1024 nodes per graph

    _Float16* Zh  = (_Float16*)d_ws;    // [M, H] f16 scratch (16 MB)
    float*    out = (float*)d_out;      // [G, N, N] f32

    dim3 blk(256);  // 8 wave32 waves

    if (H == 128 && K == 256 && (N % 64) == 0 && (M % 16) == 0) {
        const int waves1 = M / 16;
        fc1_wmma_kernel<8, 8><<<dim3((waves1 + 7) / 8), blk, 0, stream>>>(Z, W, bias, Zh, M);

        const int waves2 = G * (N >> 4) * (N >> 6);
        adj_wmma_kernel<4><<<dim3((waves2 + 7) / 8), blk, 0, stream>>>(Zh, out, G, N);
    } else {
        const long long e1 = (long long)M * H;
        fc1_generic_kernel<<<dim3((unsigned)((e1 + 255) / 256)), blk, 0, stream>>>(Z, W, bias, Zh, M, K, H);
        const long long e2 = (long long)G * N * N;
        adj_generic_kernel<<<dim3((unsigned)((e2 + 255) / 256)), blk, 0, stream>>>(Zh, out, G, N, H);
    }
}
